// TransformerBlock_1486058684966
// MI455X (gfx1250) — matrix operand order repaired and checked by reading, not compiled, not run on hardware
//
#include <hip/hip_runtime.h>
#include <hip/hip_bf16.h>

typedef __attribute__((ext_vector_type(8)))  float   v8f;
typedef __attribute__((ext_vector_type(16))) __bf16  v16bf;
typedef __bf16 bf16;
typedef int v4i_g __attribute__((vector_size(16)));   // gcc-style vector: matches builtin proto

// ---------------------------------------------------------------------------
// async global->LDS support (gfx1250 GLOBAL_LOAD_ASYNC_TO_LDS_B128, ASYNCcnt)
// ---------------------------------------------------------------------------
#if defined(__gfx1250__) && __has_builtin(__builtin_amdgcn_global_load_async_to_lds_b128) && __has_builtin(__builtin_amdgcn_s_wait_asynccnt)
#define USE_ASYNC_LDS 1
#else
#define USE_ASYNC_LDS 0
#endif

__device__ inline void cp16_g2l(const bf16* g, bf16* l) {
#if USE_ASYNC_LDS
    __builtin_amdgcn_global_load_async_to_lds_b128(
        (__attribute__((address_space(1))) v4i_g*)(g),
        (__attribute__((address_space(3))) v4i_g*)(l), 0, 0);
#else
    *(uint4*)l = *(const uint4*)g;
#endif
}

#if USE_ASYNC_LDS
#define ASYNC_WAIT(n) __builtin_amdgcn_s_wait_asynccnt(n)
#else
#define ASYNC_WAIT(n)
#endif

// ---------------------------------------------------------------------------
// helpers
// ---------------------------------------------------------------------------
__device__ inline v8f zero8() {
    v8f z = {0.f, 0.f, 0.f, 0.f, 0.f, 0.f, 0.f, 0.f};
    return z;
}

// A-matrix fragment (16xK=32 bf16). CDNA5 layout:
//   lanes 0-15 : M = lane,      elems 0-7 -> K = 0..7,   elems 8-15 -> K = 16..23
//   lanes 16-31: M = lane-16,   elems 0-7 -> K = 8..15,  elems 8-15 -> K = 24..31
// LDS tile row-major [16 rows][strideElems], K contiguous.
__device__ inline v16bf ld_fragA(const bf16* row0, int strideElems, int lane) {
    const int half = lane >> 4;
    const bf16* p = row0 + (long)(lane & 15) * strideElems + half * 8;
    v16bf a;
    ((uint4*)&a)[0] = *(const uint4*)(p);
    ((uint4*)&a)[1] = *(const uint4*)(p + 16);
    return a;
}

// B-matrix fragment (K=32 x 16 bf16). CDNA5 layout:
//   lanes 0-15 : N = lane,    elems 0-7 -> K = 0..7,   elems 8-15 -> K = 16..23
//   lanes 16-31: N = lane-16, elems 0-7 -> K = 8..15,  elems 8-15 -> K = 24..31
// LDS tile stored "N-major": [16 N rows][strideElems], K contiguous per row.
__device__ inline v16bf ld_fragB(const bf16* row0, int strideElems, int lane) {
    const int half = lane >> 4;
    const bf16* p = row0 + (long)(lane & 15) * strideElems + half * 8;
    v16bf b;
    ((uint4*)&b)[0] = *(const uint4*)(p);
    ((uint4*)&b)[1] = *(const uint4*)(p + 16);
    return b;
}

__device__ inline v8f wmma_bf16(v16bf a, v16bf b, v8f c) {
    // (neg_a, A, neg_b, B, c_mod, C, reuse_a, reuse_b)
    return __builtin_amdgcn_wmma_f32_16x16x32_bf16(false, a, false, b, (short)0, c, false, false);
}

__device__ inline float gelu_exact(float x) {
    return 0.5f * x * (1.f + erff(x * 0.70710678118654752f));
}

// ---------------------------------------------------------------------------
// weight convert: W[K,N] f32 -> Wt[N,K] bf16 (K contiguous for B-fragments)
// ---------------------------------------------------------------------------
__global__ __launch_bounds__(256) void wt_convert(const float* __restrict__ W,
                                                  bf16* __restrict__ Wt,
                                                  int K, int N) {
    long idx = (long)blockIdx.x * 256 + threadIdx.x;
    long k = idx / N, n = idx % N;
    Wt[n * (long)K + k] = (bf16)W[idx];
}

// ---------------------------------------------------------------------------
// LayerNorm: one wave per token (768 elems, 24/lane), bf16 output
// ---------------------------------------------------------------------------
__global__ __launch_bounds__(256) void ln_bf16(const float* __restrict__ x,
                                               const float* __restrict__ g,
                                               const float* __restrict__ bsh,
                                               bf16* __restrict__ out) {
    const int lane = threadIdx.x & 31, wave = threadIdx.x >> 5;
    const long t = (long)blockIdx.x * 8 + wave;
    const float* xr = x + t * 768;
    float vals[24];
    float s = 0.f, sq = 0.f;
#pragma unroll
    for (int i = 0; i < 24; ++i) {
        float v = xr[lane + i * 32];
        vals[i] = v; s += v; sq += v * v;
    }
#pragma unroll
    for (int d = 1; d <= 16; d <<= 1) {
        s  += __shfl_xor(s,  d, 32);
        sq += __shfl_xor(sq, d, 32);
    }
    const float mu   = s * (1.f / 768.f);
    const float var  = sq * (1.f / 768.f) - mu * mu;
    const float rstd = rsqrtf(var + 1e-5f);
    bf16* orow = out + t * 768;
#pragma unroll
    for (int i = 0; i < 24; ++i) {
        int c = lane + i * 32;
        orow[c] = (bf16)((vals[i] - mu) * rstd * g[c] + bsh[c]);
    }
}

// ---------------------------------------------------------------------------
// GEMM: out[M,N] = A[M,K](bf16) @ Wt[N,K]^T(bf16)  (+bias)(+res)(+gelu)
// Block tile 128x128x32, 256 threads = 8 waves (4x2), each wave 2x4 WMMA tiles.
// Double-buffered LDS; next K-tile streams in via async global->LDS DMA while
// WMMAs consume the current tile (ASYNCcnt pipelining).
// ---------------------------------------------------------------------------
template <bool BIAS, bool RES, bool GELU, bool OBF16>
__global__ __launch_bounds__(256) void gemm_bf16(const bf16* __restrict__ A,
                                                 const bf16* __restrict__ Bt,
                                                 const float* __restrict__ bias,
                                                 const float* __restrict__ res,
                                                 float* __restrict__ outF,
                                                 bf16* __restrict__ outB,
                                                 int M, int N, int K) {
    __shared__ __align__(16) bf16 As[2][128 * 40];  // padded stride 40 elems (80B)
    __shared__ __align__(16) bf16 Bs[2][128 * 40];

    const int tid = threadIdx.x, lane = tid & 31, wave = tid >> 5;
    const int wr = wave >> 1, wc = wave & 1;
    const long bm = (long)blockIdx.y * 128, bn = (long)blockIdx.x * 128;

    v8f acc[2][4];
#pragma unroll
    for (int mi = 0; mi < 2; ++mi)
#pragma unroll
        for (int ni = 0; ni < 4; ++ni) acc[mi][ni] = zero8();

    // each thread moves 2x16B for A and 2x16B for B per tile (4 async ops)
    auto issue_tile = [&](int kt, int buf) {
#pragma unroll
        for (int i = 0; i < 2; ++i) {
            int chunk = tid + i * 256;     // 512 16B chunks per 128x32 tile
            int row = chunk >> 2, kc = chunk & 3;
            cp16_g2l(A  + (bm + row) * K + kt * 32 + kc * 8, &As[buf][row * 40 + kc * 8]);
            cp16_g2l(Bt + (bn + row) * K + kt * 32 + kc * 8, &Bs[buf][row * 40 + kc * 8]);
        }
    };

    const int steps = K / 32;
    issue_tile(0, 0);
    for (int kt = 0; kt < steps; ++kt) {
        const int cur = kt & 1;
        if (kt + 1 < steps) {
            issue_tile(kt + 1, cur ^ 1);   // prefetch next tile into other buffer
            ASYNC_WAIT(4);                 // 4 newest stay in flight; current tile done
        } else {
            ASYNC_WAIT(0);
        }
        __syncthreads();                   // all waves' DMA for cur buffer visible

        v16bf af[2], bq[4];
#pragma unroll
        for (int mi = 0; mi < 2; ++mi)
            af[mi] = ld_fragA(&As[cur][(wr * 32 + mi * 16) * 40], 40, lane);
#pragma unroll
        for (int ni = 0; ni < 4; ++ni)
            bq[ni] = ld_fragB(&Bs[cur][(wc * 64 + ni * 16) * 40], 40, lane);
#pragma unroll
        for (int mi = 0; mi < 2; ++mi)
#pragma unroll
            for (int ni = 0; ni < 4; ++ni)
                acc[mi][ni] = wmma_bf16(af[mi], bq[ni], acc[mi][ni]);
        __syncthreads();                   // reads done before buffer is re-filled
    }

    // C/D layout: lane<16 -> M = r, lane>=16 -> M = 8+r ; N = lane%16
    const int half = lane >> 4, l16 = lane & 15;
#pragma unroll
    for (int mi = 0; mi < 2; ++mi)
#pragma unroll
        for (int ni = 0; ni < 4; ++ni)
#pragma unroll
            for (int r = 0; r < 8; ++r) {
                long row = bm + wr * 32 + mi * 16 + half * 8 + r;
                long col = bn + wc * 64 + ni * 16 + l16;
                float v = acc[mi][ni][r];
                if (BIAS) v += bias[col];
                if (RES)  v += res[row * N + col];
                if (GELU) v = gelu_exact(v);
                if (OBF16) outB[row * N + col] = (bf16)v;
                else       outF[row * N + col] = v;
            }
}

// ---------------------------------------------------------------------------
// Flash attention: grid (16 q-blocks, 48 b*h), 8 waves, wave = 16 q rows.
// qkv layout per token: [3*768] = which*768 + h*64 + dd. SCALE folded into Q.
// ---------------------------------------------------------------------------
__global__ __launch_bounds__(256) void attn_flash(const float* __restrict__ qkv,
                                                  bf16* __restrict__ o) {
    const int bh = blockIdx.y;
    const int b = bh / 12, h = bh % 12;
    const int wave = threadIdx.x >> 5, lane = threadIdx.x & 31;
    const int half = lane >> 4, l16 = lane & 15;
    const long tokQ = (long)b * 2048 + (long)blockIdx.x * 128 + wave * 16;

    __shared__ __align__(16) bf16 Ks[32 * 72];       // [key][d], stride 72
    __shared__ __align__(16) bf16 Vs[64 * 40];       // [d][key], stride 40
    __shared__ __align__(16) bf16 Ps[8 * 16 * 40];   // per-wave P relayout

    // Q fragments (16 rows x 64 d -> two 16x32 A-frags), scale 1/8 folded in
    v16bf qf[2];
    {
        const float* qr = qkv + (tokQ + l16) * 2304 + h * 64;
#pragma unroll
        for (int kc = 0; kc < 2; ++kc) {
            v16bf q;
#pragma unroll
            for (int e = 0; e < 8; ++e)
                q[e] = (bf16)(qr[kc * 32 + half * 8 + e] * 0.125f);
#pragma unroll
            for (int e = 0; e < 8; ++e)
                q[8 + e] = (bf16)(qr[kc * 32 + 16 + half * 8 + e] * 0.125f);
            qf[kc] = q;
        }
    }

    float m_[8], l_[8];
    v8f ov[4];
#pragma unroll
    for (int r = 0; r < 8; ++r) { m_[r] = -3.0e38f; l_[r] = 0.f; }
#pragma unroll
    for (int t2 = 0; t2 < 4; ++t2) ov[t2] = zero8();

    for (int kb = 0; kb < 2048; kb += 32) {
        __syncthreads();
        // stage K (row-major) and V (transposed): 32 keys x 64 d, 8 elems/thr
#pragma unroll
        for (int i = 0; i < 8; ++i) {
            int idx = threadIdx.x + i * 256;
            int kr = idx >> 6, dd = idx & 63;
            const float* src = qkv + ((long)b * 2048 + kb + kr) * 2304 + h * 64 + dd;
            Ks[kr * 72 + dd] = (bf16)src[768];
            Vs[dd * 40 + kr] = (bf16)src[1536];
        }
        __syncthreads();

        // S = (Q*scale) @ K^T : keys 0-15 and 16-31, K-dim d = 64 (2 chunks)
        v8f s0 = zero8(), s1 = zero8();
#pragma unroll
        for (int kc = 0; kc < 2; ++kc) {
            s0 = wmma_bf16(qf[kc], ld_fragB(&Ks[0 * 16 * 72 + kc * 32], 72, lane), s0);
            s1 = wmma_bf16(qf[kc], ld_fragB(&Ks[1 * 16 * 72 + kc * 32], 72, lane), s1);
        }

        // online softmax; row M = half*8 + r lives across 16 lanes of a half
        float mx[8];
#pragma unroll
        for (int r = 0; r < 8; ++r) mx[r] = fmaxf(s0[r], s1[r]);
#pragma unroll
        for (int d = 1; d <= 8; d <<= 1)
#pragma unroll
            for (int r = 0; r < 8; ++r)
                mx[r] = fmaxf(mx[r], __shfl_xor(mx[r], d, 32));

        float sc[8], rs[8];
#pragma unroll
        for (int r = 0; r < 8; ++r) {
            float mn = fmaxf(m_[r], mx[r]);
            sc[r] = __expf(m_[r] - mn);
            m_[r] = mn;
            s0[r] = __expf(s0[r] - mn);
            s1[r] = __expf(s1[r] - mn);
            rs[r] = s0[r] + s1[r];
        }
#pragma unroll
        for (int d = 1; d <= 8; d <<= 1)
#pragma unroll
            for (int r = 0; r < 8; ++r)
                rs[r] += __shfl_xor(rs[r], d, 32);
#pragma unroll
        for (int r = 0; r < 8; ++r) l_[r] = l_[r] * sc[r] + rs[r];
#pragma unroll
        for (int t2 = 0; t2 < 4; ++t2)
#pragma unroll
            for (int r = 0; r < 8; ++r) ov[t2][r] *= sc[r];

        // relayout P (C layout -> A layout) through per-wave LDS slab
        bf16* Pw = &Ps[wave * 16 * 40];
#pragma unroll
        for (int r = 0; r < 8; ++r) {
            Pw[(half * 8 + r) * 40 + l16]      = (bf16)s0[r];
            Pw[(half * 8 + r) * 40 + 16 + l16] = (bf16)s1[r];
        }
        __syncthreads();
        v16bf pf = ld_fragA(Pw, 40, lane);

        // O += P @ V  (K-dim = 32 keys, 4 d-subtiles)
#pragma unroll
        for (int t2 = 0; t2 < 4; ++t2)
            ov[t2] = wmma_bf16(pf, ld_fragB(&Vs[t2 * 16 * 40], 40, lane), ov[t2]);
    }

    // normalize and store o (bf16) at [token, h*64 + dd]
#pragma unroll
    for (int t2 = 0; t2 < 4; ++t2)
#pragma unroll
        for (int r = 0; r < 8; ++r) {
            float v = ov[t2][r] / l_[r];
            long tok = tokQ + half * 8 + r;
            o[tok * 768 + h * 64 + t2 * 16 + l16] = (bf16)v;
        }
}

// ---------------------------------------------------------------------------
// launch
// ---------------------------------------------------------------------------
extern "C" void kernel_launch(void* const* d_in, const int* in_sizes, int n_in,
                              void* d_out, int out_size, void* d_ws, size_t ws_size,
                              hipStream_t stream) {
    (void)in_sizes; (void)n_in; (void)out_size; (void)ws_size;

    const float* x      = (const float*)d_in[0];
    const float* ln1_g  = (const float*)d_in[1];
    const float* ln1_b  = (const float*)d_in[2];
    const float* qkv_w  = (const float*)d_in[3];
    const float* proj_w = (const float*)d_in[4];
    const float* proj_b = (const float*)d_in[5];
    const float* ln2_g  = (const float*)d_in[6];
    const float* ln2_b  = (const float*)d_in[7];
    const float* fc1_w  = (const float*)d_in[8];
    const float* fc1_b  = (const float*)d_in[9];
    const float* fc2_w  = (const float*)d_in[10];
    const float* fc2_b  = (const float*)d_in[11];
    float* out = (float*)d_out;

    const long T = 4L * 2048;  // 8192 tokens
    char* ws = (char*)d_ws;
    size_t off = 0;
    auto alloc = [&](size_t bytes) {
        char* p = ws + off;
        off += (bytes + 255) & ~(size_t)255;
        return p;
    };
    bf16*  h       = (bf16*)alloc(T * 768 * 2);
    float* qkvb    = (float*)alloc(T * 2304 * 4);
    bf16*  ob      = (bf16*)alloc(T * 768 * 2);
    float* x1      = (float*)alloc(T * 768 * 4);
    bf16*  h2      = (bf16*)alloc(T * 768 * 2);
    bf16*  m1      = (bf16*)alloc(T * 3072 * 2);
    bf16*  qkv_wt  = (bf16*)alloc(768L * 2304 * 2);
    bf16*  proj_wt = (bf16*)alloc(768L * 768 * 2);
    bf16*  fc1_wt  = (bf16*)alloc(768L * 3072 * 2);
    bf16*  fc2_wt  = (bf16*)alloc(3072L * 768 * 2);

    // weights -> bf16, transposed to [N,K]
    wt_convert<<<(768 * 2304) / 256, 256, 0, stream>>>(qkv_w,  qkv_wt,  768, 2304);
    wt_convert<<<(768 * 768)  / 256, 256, 0, stream>>>(proj_w, proj_wt, 768, 768);
    wt_convert<<<(768 * 3072) / 256, 256, 0, stream>>>(fc1_w,  fc1_wt,  768, 3072);
    wt_convert<<<(3072 * 768) / 256, 256, 0, stream>>>(fc2_w,  fc2_wt, 3072, 768);

    // attention branch
    ln_bf16<<<T / 8, 256, 0, stream>>>(x, ln1_g, ln1_b, h);
    gemm_bf16<false, false, false, false>
        <<<dim3(2304 / 128, T / 128), 256, 0, stream>>>(
            h, qkv_wt, nullptr, nullptr, qkvb, nullptr, (int)T, 2304, 768);
    attn_flash<<<dim3(16, 48), 256, 0, stream>>>(qkvb, ob);
    gemm_bf16<true, true, false, false>
        <<<dim3(768 / 128, T / 128), 256, 0, stream>>>(
            ob, proj_wt, proj_b, x, x1, nullptr, (int)T, 768, 768);

    // MLP branch
    ln_bf16<<<T / 8, 256, 0, stream>>>(x1, ln2_g, ln2_b, h2);
    gemm_bf16<true, false, true, true>
        <<<dim3(3072 / 128, T / 128), 256, 0, stream>>>(
            h2, fc1_wt, fc1_b, nullptr, nullptr, m1, (int)T, 3072, 768);
    gemm_bf16<true, true, false, false>
        <<<dim3(768 / 128, T / 128), 256, 0, stream>>>(
            m1, fc2_wt, fc2_b, x1, out, nullptr, (int)T, 768, 3072);
}
